// ChemicalSpecialist2D_35914516529887
// MI455X (gfx1250) — matrix operand order repaired and checked by reading, not compiled, not run on hardware
//
#include <hip/hip_runtime.h>
#include <cstdint>
#include <cstddef>

typedef __attribute__((ext_vector_type(16))) _Float16 v16h;
typedef __attribute__((ext_vector_type(8)))  float    v8f;

#define N_NODES 20000
#define N_EDGES 320000
#define HID     256
#define BDIM    64
#define KMSG    576            // 2*HID + BDIM
#define MT      32             // edges per block
#define AST     (KMSG + 8)     // A-tile row stride in f16 (584 -> 1168B, 16B aligned)
#define MST     (HID + 8)      // msg-tile row stride in f16 (264 -> 528B)
#define KT1     (KMSG / 32)    // 18 K-steps for GEMM1
#define KT2     (HID / 32)     // 8 K-steps for GEMM2
#define NTILES  (HID / 16)     // 16 column tiles
#define FRAGDW  256            // dwords per B fragment (32 lanes x 8)

static __device__ __forceinline__ int clampi(int v, int lo, int hi) {
    return v < lo ? lo : (v > hi ? hi : v);
}

// ---------------------------------------------------------------------------
// Weight repack: [K][N] fp32 row-major -> per-(ktile,ntile) WMMA B fragments.
// B fragment (f16 32x16): lane holds column n = base + (lane&15);
// dword v packs K pair ((v>>2)*16 + half*8 + 2*(v&3), +1) (table order).
// Stored as 8 consecutive dwords per lane -> coalesced.
// ---------------------------------------------------------------------------
__global__ void k_pack_b(const float* __restrict__ w, uint32_t* __restrict__ dst,
                         int kdim, int ktiles) {
    int bid  = blockIdx.x;                 // l*ktiles*16 + kt*16 + nt
    int nt   = bid & 15;
    int kt   = (bid >> 4) % ktiles;
    int l    = (bid >> 4) / ktiles;
    int lane = threadIdx.x;
    int n    = nt * 16 + (lane & 15);
    int half = lane >> 4;
    const float* W = w + (size_t)l * kdim * HID;
    uint32_t* o = dst + ((size_t)bid * 32 + lane) * 8;
#pragma unroll
    for (int v = 0; v < 8; ++v) {
        int k = kt * 32 + (v >> 2) * 16 + half * 8 + 2 * (v & 3);
        union { _Float16 f[2]; uint32_t u; } p;
        p.f[0] = (_Float16)W[(size_t)k * HID + n];
        p.f[1] = (_Float16)W[(size_t)(k + 1) * HID + n];
        o[v] = p.u;
    }
}

// ---------------------------------------------------------------------------
// Node embedding: concat(atom[64], charge[32], hybrid[32]) padded to 256.
// ---------------------------------------------------------------------------
__global__ void k_embed(const float* __restrict__ x,
                        const float* __restrict__ at_t,
                        const float* __restrict__ ch_t,
                        const float* __restrict__ hy_t,
                        float* __restrict__ h) {
    int i = blockIdx.x * 256 + threadIdx.x;      // N*256 threads
    int node = i >> 8, f = i & 255;
    int a  = clampi((int)x[node * 3 + 0], 0, 10);
    int hy = clampi((int)x[node * 3 + 1], 0, 7);
    int ch = clampi((int)x[node * 3 + 2] + 3, 0, 6);
    float v = 0.f;
    if (f < 64)       v = at_t[a * 64 + f];
    else if (f < 96)  v = ch_t[ch * 32 + (f - 64)];
    else if (f < 128) v = hy_t[hy * 32 + (f - 96)];
    h[i] = v;
}

// h16 = f16(h_in); h_out = h_in  (double buffer init for segment_sum)
__global__ void k_h16(const float* __restrict__ hin, _Float16* __restrict__ h16,
                      float* __restrict__ hout) {
    size_t i = (size_t)blockIdx.x * 256 + threadIdx.x;
    float v = hin[i];
    h16[i]  = (_Float16)v;
    hout[i] = v;
}

// ---------------------------------------------------------------------------
// Fused edge MLP: gather -> GEMM1(576x256) -> ReLU -> GEMM2(256x256)
//                 -> *sigmoid gate -> atomic segment_sum into h_out.
// Block = 32 edges, 8 wave32s.  Wave w owns column tiles {2w, 2w+1} across
// BOTH 16-row strips: every B fragment load is unique within the block
// (halves L2 B-traffic), and each A/B fragment feeds 2 WMMAs.
// ---------------------------------------------------------------------------
__global__ __launch_bounds__(256) void k_edge(
    const _Float16* __restrict__ h16, float* __restrict__ hout,
    const int* __restrict__ ei, const float* __restrict__ eattr,
    const float* __restrict__ bond_table,
    const float* __restrict__ attw, const float* __restrict__ attb,
    const uint32_t* __restrict__ w1p, const float* __restrict__ b1,
    const uint32_t* __restrict__ w2p, const float* __restrict__ b2,
    int layer) {
    __shared__ _Float16 Atile[MT][AST];
    __shared__ _Float16 Mtile[MT][MST];
    __shared__ int   rowS[MT], colS[MT], btS[MT];
    __shared__ float gateS[MT];
    __shared__ float gpart[MT][4];

    const int tid = threadIdx.x;
    const int e0  = blockIdx.x * MT;

    if (tid < MT) {
        int e = e0 + tid;
        rowS[tid] = ei[e];
        colS[tid] = ei[N_EDGES + e];
        btS[tid]  = clampi((int)eattr[e], 0, 4);
    }
    __syncthreads();

    // Stage h16[row] and h16[col]: 32 edges x 2 x 128 dwords, coalesced.
    {
        const uint32_t* hsrc = (const uint32_t*)h16;
        for (int it = 0; it < (MT * 2 * 128) / 256; ++it) {
            int idx = tid + it * 256;
            int m = idx >> 8, w = idx & 255;
            int src = w >> 7, d = w & 127;
            int node = src ? colS[m] : rowS[m];
            ((uint32_t*)&Atile[m][0])[src * 128 + d] = hsrc[(size_t)node * 128 + d];
        }
    }
    // Bond embedding -> columns 512..575
    for (int it = 0; it < (MT * BDIM) / 256; ++it) {
        int idx = tid + it * 256;
        int m = idx >> 6, j = idx & 63;
        Atile[m][2 * HID + j] = (_Float16)bond_table[btS[m] * BDIM + j];
    }
    // Gate: sigmoid(bond_emb . att_w + att_b), 4 partial lanes per edge.
    if (tid < MT * 4) {
        int m = tid >> 2, p = tid & 3;
        const float* br = bond_table + btS[m] * BDIM;
        const float* aw = attw + layer * BDIM;
        float s = 0.f;
#pragma unroll
        for (int j = 0; j < 16; ++j) s += br[p * 16 + j] * aw[p * 16 + j];
        gpart[m][p] = s;
    }
    __syncthreads();
    if (tid < MT) {
        float g = gpart[tid][0] + gpart[tid][1] + gpart[tid][2] + gpart[tid][3]
                + attb[layer];
        gateS[tid] = 1.f / (1.f + __expf(-g));
    }
    __syncthreads();

    const int wave = tid >> 5, lane = tid & 31;
    const int half = lane >> 4;
    const int mlo  = lane & 15;              // A-frag row within a strip

    // per-lane LDS row pointers for the two 16-row strips
    const uint32_t* arow0 = (const uint32_t*)&Atile[mlo][0];
    const uint32_t* arow1 = (const uint32_t*)&Atile[16 + mlo][0];

    // per-wave B base: column tiles 2w and 2w+1; all kt displacements are
    // compile-time constants -> folded into global_load immediates.
    const uint32_t* b1base = w1p + (size_t)layer * KT1 * NTILES * FRAGDW
                                 + (size_t)(2 * wave) * FRAGDW + lane * 8;
    const uint32_t* b2base = w2p + (size_t)layer * KT2 * NTILES * FRAGDW
                                 + (size_t)(2 * wave) * FRAGDW + lane * 8;

    // ---------------- GEMM1: [32 x 576] @ [576 x 256] ----------------
    v8f acc00 = {}, acc01 = {}, acc10 = {}, acc11 = {};
#pragma unroll
    for (int kt = 0; kt < KT1; ++kt) {
        union { uint32_t u[8]; v16h h; } A0, A1, B0, B1;
        const int kb2 = kt * 16 + half * 4;  // dword base in A row
#pragma unroll
        for (int v = 0; v < 4; ++v) {
            A0.u[v]     = arow0[kb2 + v];
            A0.u[4 + v] = arow0[kb2 + 8 + v];
            A1.u[v]     = arow1[kb2 + v];
            A1.u[4 + v] = arow1[kb2 + 8 + v];
        }
#pragma unroll
        for (int v = 0; v < 8; ++v) {
            B0.u[v] = b1base[kt * (NTILES * FRAGDW) + v];
            B1.u[v] = b1base[kt * (NTILES * FRAGDW) + FRAGDW + v];
        }
        acc00 = __builtin_amdgcn_wmma_f32_16x16x32_f16(false, A0.h, false, B0.h,
                                                       (short)0, acc00, false, false);
        acc01 = __builtin_amdgcn_wmma_f32_16x16x32_f16(false, A0.h, false, B1.h,
                                                       (short)0, acc01, false, false);
        acc10 = __builtin_amdgcn_wmma_f32_16x16x32_f16(false, A1.h, false, B0.h,
                                                       (short)0, acc10, false, false);
        acc11 = __builtin_amdgcn_wmma_f32_16x16x32_f16(false, A1.h, false, B1.h,
                                                       (short)0, acc11, false, false);
    }
    // bias + ReLU -> Mtile (f16); C/D layout: half 0 -> M=r, half 1 -> M=8+r
    {
        const int n0 = (2 * wave) * 16 + mlo;
        const int n1 = n0 + 16;
        const float bias0 = b1[n0], bias1 = b1[n1];
#pragma unroll
        for (int r = 0; r < 8; ++r) {
            int m0 = half * 8 + r;           // strip 0 rows
            int m1 = 16 + half * 8 + r;      // strip 1 rows
            float v;
            v = acc00[r] + bias0; Mtile[m0][n0] = (_Float16)(v > 0.f ? v : 0.f);
            v = acc01[r] + bias1; Mtile[m0][n1] = (_Float16)(v > 0.f ? v : 0.f);
            v = acc10[r] + bias0; Mtile[m1][n0] = (_Float16)(v > 0.f ? v : 0.f);
            v = acc11[r] + bias1; Mtile[m1][n1] = (_Float16)(v > 0.f ? v : 0.f);
        }
    }
    __syncthreads();

    // ---------------- GEMM2: [32 x 256] @ [256 x 256] ----------------
    v8f d00 = {}, d01 = {}, d10 = {}, d11 = {};
    const uint32_t* mrow0 = (const uint32_t*)&Mtile[mlo][0];
    const uint32_t* mrow1 = (const uint32_t*)&Mtile[16 + mlo][0];
#pragma unroll
    for (int kt = 0; kt < KT2; ++kt) {
        union { uint32_t u[8]; v16h h; } A0, A1, B0, B1;
        const int kb2 = kt * 16 + half * 4;
#pragma unroll
        for (int v = 0; v < 4; ++v) {
            A0.u[v]     = mrow0[kb2 + v];
            A0.u[4 + v] = mrow0[kb2 + 8 + v];
            A1.u[v]     = mrow1[kb2 + v];
            A1.u[4 + v] = mrow1[kb2 + 8 + v];
        }
#pragma unroll
        for (int v = 0; v < 8; ++v) {
            B0.u[v] = b2base[kt * (NTILES * FRAGDW) + v];
            B1.u[v] = b2base[kt * (NTILES * FRAGDW) + FRAGDW + v];
        }
        d00 = __builtin_amdgcn_wmma_f32_16x16x32_f16(false, A0.h, false, B0.h,
                                                     (short)0, d00, false, false);
        d01 = __builtin_amdgcn_wmma_f32_16x16x32_f16(false, A0.h, false, B1.h,
                                                     (short)0, d01, false, false);
        d10 = __builtin_amdgcn_wmma_f32_16x16x32_f16(false, A1.h, false, B0.h,
                                                     (short)0, d10, false, false);
        d11 = __builtin_amdgcn_wmma_f32_16x16x32_f16(false, A1.h, false, B1.h,
                                                     (short)0, d11, false, false);
    }
    // bias + gate + atomic scatter (segment_sum over rows)
    {
        const int n0 = (2 * wave) * 16 + mlo;
        const int n1 = n0 + 16;
        const float bias0 = b2[n0], bias1 = b2[n1];
#pragma unroll
        for (int r = 0; r < 8; ++r) {
            int m0 = half * 8 + r;
            int m1 = 16 + half * 8 + r;
            float g0 = gateS[m0], g1 = gateS[m1];
            atomicAdd(&hout[(size_t)rowS[m0] * HID + n0], (d00[r] + bias0) * g0);
            atomicAdd(&hout[(size_t)rowS[m0] * HID + n1], (d01[r] + bias1) * g0);
            atomicAdd(&hout[(size_t)rowS[m1] * HID + n0], (d10[r] + bias0) * g1);
            atomicAdd(&hout[(size_t)rowS[m1] * HID + n1], (d11[r] + bias1) * g1);
        }
    }
}

// ---------------------------------------------------------------------------
// chemical_properties MLP, fp32 exact: one node per block.
// w1 is [256][128] row-major -> threads (o) read contiguous addresses per k.
// ---------------------------------------------------------------------------
__global__ __launch_bounds__(256) void k_props(
    const float* __restrict__ h, const float* __restrict__ w1,
    const float* __restrict__ b1, const float* __restrict__ w2,
    const float* __restrict__ b2, float* __restrict__ out) {
    __shared__ float hrow[HID];
    __shared__ float tmp[128];
    int n = blockIdx.x, tid = threadIdx.x;
    hrow[tid] = h[(size_t)n * HID + tid];
    __syncthreads();
    if (tid < 128) {
        float s = b1[tid];
        for (int k = 0; k < HID; ++k) s += hrow[k] * w1[k * 128 + tid];
        tmp[tid] = s > 0.f ? s : 0.f;
    }
    __syncthreads();
    if (tid < 32) {
        float s = b2[tid];
        for (int k = 0; k < 128; ++k) s += tmp[k] * w2[k * 32 + tid];
        out[(size_t)n * 32 + tid] = s;
    }
}

__global__ void k_zero(float* p, int n) {
    int i = blockIdx.x * 256 + threadIdx.x;
    if (i < n) p[i] = 0.f;
}

__global__ __launch_bounds__(256) void k_pool(const float* __restrict__ h,
                                              float* __restrict__ pooled) {
    int c = threadIdx.x;
    int r0 = blockIdx.x * (N_NODES / 100);   // 100 blocks x 200 rows
    float s = 0.f;
    for (int r = 0; r < N_NODES / 100; ++r) s += h[(size_t)(r0 + r) * HID + c];
    atomicAdd(&pooled[c], s);
}

__global__ void k_fg(const float* __restrict__ pooled, const float* __restrict__ fgw,
                     const float* __restrict__ fgb, float* __restrict__ fg) {
    int t = threadIdx.x;                      // 64 threads
    int kk = t >> 4, o = t & 15;
    const float* W = fgw + (size_t)kk * HID * 16;
    float s = fgb[t];
    const float inv = 1.f / (float)N_NODES;
    for (int d = 0; d < HID; ++d) s += (pooled[d] * inv) * W[d * 16 + o];
    fg[t] = s;
}

__global__ void k_bcast(const float* __restrict__ fg, float* __restrict__ out) {
    int i = blockIdx.x * 256 + threadIdx.x;
    if (i < N_NODES * 64) out[i] = fg[i & 63];
}

__global__ void k_copy_h(const float* __restrict__ h, float* __restrict__ out) {
    size_t i = (size_t)blockIdx.x * 256 + threadIdx.x;
    out[i] = h[i];
}

__global__ void k_atoms(const float* __restrict__ x, float* __restrict__ out) {
    int i = blockIdx.x * 256 + threadIdx.x;
    if (i < N_NODES) out[i] = (float)clampi((int)x[i * 3], 0, 10);
}

// ---------------------------------------------------------------------------
extern "C" void kernel_launch(void* const* d_in, const int* in_sizes, int n_in,
                              void* d_out, int out_size, void* d_ws, size_t ws_size,
                              hipStream_t stream) {
    const float* x        = (const float*)d_in[0];
    const int*   ei       = (const int*)d_in[1];
    const float* eattr    = (const float*)d_in[2];
    const float* at_t     = (const float*)d_in[4];
    const float* ch_t     = (const float*)d_in[5];
    const float* hy_t     = (const float*)d_in[6];
    const float* bond_t   = (const float*)d_in[7];
    const float* gnn_w1   = (const float*)d_in[8];
    const float* gnn_b1   = (const float*)d_in[9];
    const float* gnn_w2   = (const float*)d_in[10];
    const float* gnn_b2   = (const float*)d_in[11];
    const float* attw     = (const float*)d_in[12];
    const float* attb     = (const float*)d_in[13];
    const float* pw1      = (const float*)d_in[14];
    const float* pb1      = (const float*)d_in[15];
    const float* pw2      = (const float*)d_in[16];
    const float* pb2      = (const float*)d_in[17];
    const float* fgw      = (const float*)d_in[18];
    const float* fgb      = (const float*)d_in[19];
    float* out = (float*)d_out;

    // workspace carve-up (~53 MB)
    char* w = (char*)d_ws;
    float*     hA     = (float*)w;      w += (size_t)N_NODES * HID * 4;
    float*     hB     = (float*)w;      w += (size_t)N_NODES * HID * 4;
    _Float16*  h16    = (_Float16*)w;   w += (size_t)N_NODES * HID * 2;
    uint32_t*  w1p    = (uint32_t*)w;   w += (size_t)3 * KT1 * NTILES * 32 * 8 * 4;
    uint32_t*  w2p    = (uint32_t*)w;   w += (size_t)3 * KT2 * NTILES * 32 * 8 * 4;
    float*     pooled = (float*)w;      w += 256 * 4;
    float*     fgbuf  = (float*)w;      w += 64 * 4;

    const size_t OFF_PROP = (size_t)N_NODES * HID;
    const size_t OFF_FG   = OFF_PROP + (size_t)N_NODES * 32;
    const size_t OFF_AT   = OFF_FG + (size_t)N_NODES * 64;

    k_pack_b<<<3 * KT1 * NTILES, 32, 0, stream>>>(gnn_w1, w1p, KMSG, KT1);
    k_pack_b<<<3 * KT2 * NTILES, 32, 0, stream>>>(gnn_w2, w2p, HID, KT2);
    k_embed<<<N_NODES, 256, 0, stream>>>(x, at_t, ch_t, hy_t, hA);

    float* hin = hA;
    float* hout = hB;
    for (int l = 0; l < 3; ++l) {
        k_h16<<<N_NODES, 256, 0, stream>>>(hin, h16, hout);
        k_edge<<<N_EDGES / MT, 256, 0, stream>>>(
            h16, hout, ei, eattr, bond_t, attw, attb,
            w1p, gnn_b1 + l * HID, w2p, gnn_b2 + l * HID, l);
        float* t = hin; hin = hout; hout = t;
    }
    float* hfin = hin;   // = hB after 3 layers

    k_copy_h<<<N_NODES, 256, 0, stream>>>(hfin, out);
    k_props<<<N_NODES, 256, 0, stream>>>(hfin, pw1, pb1, pw2, pb2, out + OFF_PROP);
    k_zero<<<1, 256, 0, stream>>>(pooled, 256);
    k_pool<<<100, 256, 0, stream>>>(hfin, pooled);
    k_fg<<<1, 64, 0, stream>>>(pooled, fgw, fgb, fgbuf);
    k_bcast<<<(N_NODES * 64 + 255) / 256, 256, 0, stream>>>(fgbuf, out + OFF_FG);
    k_atoms<<<(N_NODES + 255) / 256, 256, 0, stream>>>(x, out + OFF_AT);
}
